// IO_B_18562848653592
// MI455X (gfx1250) — matrix operand order repaired and checked by reading, not compiled, not run on hardware
//
#include <hip/hip_runtime.h>

// ---------------------------------------------------------------------------
// Sizes from the reference
// ---------------------------------------------------------------------------
#define BB   32
#define NN   10000
#define DIM  64
#define HID  128
#define TILES 625              // NN / 16
#define GY   16                // workgroups per batch
#define WPB  4                 // waves per block (128 threads)
#define NWAVES (GY * WPB)      // waves per batch = 64
#define TPW  ((TILES + NWAVES - 1) / NWAVES)   // tiles per wave = 10
#define LDS_STRIDE 136         // 128 + 8 half padding -> conflict-free b128 reads

typedef __attribute__((ext_vector_type(16))) _Float16 v16h;
typedef __attribute__((ext_vector_type(8)))  float    v8f;

union FragH { v16h h; uint4 u[2]; };
union Pack4 { _Float16 h[4]; uint2 u; };

// ---------------------------------------------------------------------------
// Kernel 0: convert W1/W2 (fp32, row-major [HID][HID]) into f16 B-fragment
// layout for v_wmma_f32_16x16x32_f16, and zero the s accumulator.
// Fragment layout: frag[(nt*4+ks)][lane][j] = W[n][k],
//   n = nt*16 + lane%16,  k = ks*32 + (lane/16)*16 + j   (j = 0..15)
// a contiguous k order (not the ISA table order), the same one the A-fragment loads use.
// ---------------------------------------------------------------------------
__global__ void prep_kernel(const float* __restrict__ w1,
                            const float* __restrict__ w2,
                            _Float16* __restrict__ w1f,
                            _Float16* __restrict__ w2f,
                            float* __restrict__ s)
{
    int tid    = blockIdx.x * blockDim.x + threadIdx.x;
    int stride = gridDim.x * blockDim.x;
    for (int idx = tid; idx < HID * HID; idx += stride) {
        int j    = idx & 15;
        int lane = (idx >> 4) & 31;
        int ks   = (idx >> 9) & 3;
        int nt   = idx >> 11;
        int n    = nt * 16 + (lane & 15);
        int k    = ks * 32 + (lane >> 4) * 16 + j;
        w1f[idx] = (_Float16)w1[n * HID + k];
        w2f[idx] = (_Float16)w2[n * HID + k];
    }
    for (int idx = tid; idx < BB * HID; idx += stride) s[idx] = 0.0f;
}

// ---------------------------------------------------------------------------
// Kernel 1: fused per-node MLP + weighted reduction over nodes.
//   per 16-node tile:  c = [x_n | x_starter] (f16, LDS)
//   h1 = relu(c @ W1^T + b1) ; h2 = relu(h1 @ W2^T + b2)
//   s[b,:] += sum_n h2[n,:] * adj[n]
// ---------------------------------------------------------------------------
__global__ void __launch_bounds__(128)
main_kernel(const float* __restrict__ x,
            const float* __restrict__ adj,
            const float* __restrict__ b1,
            const float* __restrict__ b2,
            const _Float16* __restrict__ w1f,
            const _Float16* __restrict__ w2f,
            float* __restrict__ s,
            const int* __restrict__ ip)
{
    __shared__ __attribute__((aligned(16))) _Float16 lds[WPB][2][16][LDS_STRIDE];

    const int lane = threadIdx.x & 31;
    const int wid  = threadIdx.x >> 5;
    const int b    = blockIdx.x;
    const int col  = lane & 15;   // row for A loads / column for C,D tiles
    const int hi   = lane >> 4;   // lane-group select
    const int iNode = ip[0];

    _Float16 (*ctile)[LDS_STRIDE] = lds[wid][0];
    _Float16 (*htile)[LDS_STRIDE] = lds[wid][1];

    // Per-lane bias values for each 16-column output tile
    float b1v[8], b2v[8];
#pragma unroll
    for (int nt = 0; nt < 8; ++nt) {
        b1v[nt] = b1[nt * 16 + col];
        b2v[nt] = b2[nt * 16 + col];
    }

    // Write the broadcast starter half (cols 64..127) of the c-tile once.
    {
        const float* st = x + ((size_t)b * NN + iNode) * DIM + hi * 32;
#pragma unroll
        for (int q = 0; q < 8; ++q) {
            float4 f = *(const float4*)(st + q * 4);
            Pack4 p;
            p.h[0] = (_Float16)f.x; p.h[1] = (_Float16)f.y;
            p.h[2] = (_Float16)f.z; p.h[3] = (_Float16)f.w;
            *(uint2*)&ctile[col][DIM + hi * 32 + q * 4] = p.u;
        }
    }

    float sum[8][8];
#pragma unroll
    for (int nt = 0; nt < 8; ++nt)
#pragma unroll
        for (int v = 0; v < 8; ++v) sum[nt][v] = 0.0f;

    const int wgl = blockIdx.y * WPB + wid;
    const int t0  = wgl * TPW;
    const int t1  = (t0 + TPW < TILES) ? t0 + TPW : TILES;

    for (int t = t0; t < t1; ++t) {
        const int nodeBase = t * 16;

        // Stage x rows (cols 0..63) into the c-tile as f16.
        const float* xr = x + ((size_t)b * NN + nodeBase + col) * DIM + hi * 32;
        if (t + 1 < t1) __builtin_prefetch(xr + 16 * DIM, 0, 1);
#pragma unroll
        for (int q = 0; q < 8; ++q) {
            float4 f = *(const float4*)(xr + q * 4);
            Pack4 p;
            p.h[0] = (_Float16)f.x; p.h[1] = (_Float16)f.y;
            p.h[2] = (_Float16)f.z; p.h[3] = (_Float16)f.w;
            *(uint2*)&ctile[col][hi * 32 + q * 4] = p.u;
        }

        // adj values for the rows this lane's accumulators cover (M = v + 8*hi)
        float adjv[8];
        {
            const float* ap = adj + nodeBase + hi * 8;
            float4 a0 = *(const float4*)(ap);
            float4 a1 = *(const float4*)(ap + 4);
            adjv[0] = a0.x; adjv[1] = a0.y; adjv[2] = a0.z; adjv[3] = a0.w;
            adjv[4] = a1.x; adjv[5] = a1.y; adjv[6] = a1.z; adjv[7] = a1.w;
        }

        // ---- GEMM1: h1 = c @ W1^T ----
        v8f acc[8] = {};
#pragma unroll
        for (int ks = 0; ks < 4; ++ks) {
            FragH a;
            const _Float16* ar = &ctile[col][ks * 32 + hi * 16];
            a.u[0] = *(const uint4*)(ar);
            a.u[1] = *(const uint4*)(ar + 8);
#pragma unroll
            for (int nt = 0; nt < 8; ++nt) {
                FragH bf;
                const _Float16* bp = w1f + (((nt * 4 + ks) * 32 + lane) << 4);
                bf.u[0] = *(const uint4*)(bp);
                bf.u[1] = *(const uint4*)(bp + 8);
                acc[nt] = __builtin_amdgcn_wmma_f32_16x16x32_f16(
                    false, a.h, false, bf.h, (short)0, acc[nt], false, false);
            }
        }

        // relu + bias -> h-tile (f16) ; D layout: M = v + 8*hi, N = nt*16 + col
#pragma unroll
        for (int nt = 0; nt < 8; ++nt)
#pragma unroll
            for (int v = 0; v < 8; ++v) {
                float val = acc[nt][v] + b1v[nt];
                val = val > 0.0f ? val : 0.0f;
                htile[v + hi * 8][nt * 16 + col] = (_Float16)val;
            }

        // ---- GEMM2: h2 = h1 @ W2^T ----
        v8f acc2[8] = {};
#pragma unroll
        for (int ks = 0; ks < 4; ++ks) {
            FragH a;
            const _Float16* ar = &htile[col][ks * 32 + hi * 16];
            a.u[0] = *(const uint4*)(ar);
            a.u[1] = *(const uint4*)(ar + 8);
#pragma unroll
            for (int nt = 0; nt < 8; ++nt) {
                FragH bf;
                const _Float16* bp = w2f + (((nt * 4 + ks) * 32 + lane) << 4);
                bf.u[0] = *(const uint4*)(bp);
                bf.u[1] = *(const uint4*)(bp + 8);
                acc2[nt] = __builtin_amdgcn_wmma_f32_16x16x32_f16(
                    false, a.h, false, bf.h, (short)0, acc2[nt], false, false);
            }
        }

        // relu + bias, scale by adj[node], accumulate
#pragma unroll
        for (int nt = 0; nt < 8; ++nt)
#pragma unroll
            for (int v = 0; v < 8; ++v) {
                float val = acc2[nt][v] + b2v[nt];
                val = val > 0.0f ? val : 0.0f;
                sum[nt][v] += val * adjv[v];
            }
    }

    // Reduce this wave's partials into s[b, :]
#pragma unroll
    for (int nt = 0; nt < 8; ++nt) {
        float p = sum[nt][0] + sum[nt][1] + sum[nt][2] + sum[nt][3]
                + sum[nt][4] + sum[nt][5] + sum[nt][6] + sum[nt][7];
        __hip_atomic_fetch_add(&s[b * HID + nt * 16 + col], p,
                               __ATOMIC_RELAXED, __HIP_MEMORY_SCOPE_AGENT);
    }
}

// ---------------------------------------------------------------------------
// Kernel 2: tiny tail.  y = relu(s@W3^T+b3); y = relu(y@W4^T+b4);
//           out = [starter | y] @ W5^T + b5        (32x64 fp32 output)
// ---------------------------------------------------------------------------
__global__ void __launch_bounds__(128)
tail_kernel(const float* __restrict__ x,
            const float* __restrict__ s,
            const float* __restrict__ w3, const float* __restrict__ b3,
            const float* __restrict__ w4, const float* __restrict__ b4,
            const float* __restrict__ w5, const float* __restrict__ b5,
            float* __restrict__ out,
            const int* __restrict__ ip)
{
    __shared__ float y1[BB][HID];
    __shared__ float y2[BB][HID];
    const int t = threadIdx.x;       // 0..127, one output column per thread
    const int iNode = ip[0];

    for (int r = 0; r < BB; ++r) {
        float acc = b3[t];
        for (int k = 0; k < HID; ++k) acc += s[r * HID + k] * w3[t * HID + k];
        y1[r][t] = acc > 0.0f ? acc : 0.0f;
    }
    __syncthreads();
    for (int r = 0; r < BB; ++r) {
        float acc = b4[t];
        for (int k = 0; k < HID; ++k) acc += y1[r][k] * w4[t * HID + k];
        y2[r][t] = acc > 0.0f ? acc : 0.0f;
    }
    __syncthreads();
    if (t < DIM) {
        for (int r = 0; r < BB; ++r) {
            float acc = b5[t];
            const float* st = x + ((size_t)r * NN + iNode) * DIM;
            for (int k = 0; k < DIM; ++k) acc += st[k] * w5[t * (DIM + HID) + k];
            for (int k = 0; k < HID; ++k) acc += y2[r][k] * w5[t * (DIM + HID) + DIM + k];
            out[r * DIM + t] = acc;
        }
    }
}

// ---------------------------------------------------------------------------
extern "C" void kernel_launch(void* const* d_in, const int* in_sizes, int n_in,
                              void* d_out, int out_size, void* d_ws, size_t ws_size,
                              hipStream_t stream)
{
    const float* x   = (const float*)d_in[0];
    const float* adj = (const float*)d_in[1];
    const float* w1  = (const float*)d_in[2];   // n2e_w
    const float* b1  = (const float*)d_in[3];
    const float* w2  = (const float*)d_in[4];   // e2e_w
    const float* b2  = (const float*)d_in[5];
    const float* w3  = (const float*)d_in[6];   // e2n_w
    const float* b3  = (const float*)d_in[7];
    const float* w4  = (const float*)d_in[8];   // n2n_w
    const float* b4  = (const float*)d_in[9];
    const float* w5  = (const float*)d_in[10];  // out_w
    const float* b5  = (const float*)d_in[11];
    const int*   ip  = (const int*)d_in[12];
    float* out = (float*)d_out;

    char* ws = (char*)d_ws;
    _Float16* w1f = (_Float16*)(ws);                 // 32 KB
    _Float16* w2f = (_Float16*)(ws + 32768);         // 32 KB
    float*    s   = (float*)(ws + 65536);            // 16 KB

    prep_kernel<<<64, 256, 0, stream>>>(w1, w2, w1f, w2f, s);
    main_kernel<<<dim3(BB, GY), WPB * 32, 0, stream>>>(x, adj, b1, b2, w1f, w2f, s, ip);
    tail_kernel<<<1, 128, 0, stream>>>(x, s, w3, b3, w4, b4, w5, b5, out, ip);
}
